// PairEmbed_75728863363411
// MI455X (gfx1250) — matrix operand order repaired and checked by reading, not compiled, not run on hardware
//
#include <hip/hip_runtime.h>
#include <math.h>

typedef _Float16 v8h  __attribute__((ext_vector_type(8)));
typedef _Float16 v16h __attribute__((ext_vector_type(16)));
typedef float    v8f  __attribute__((ext_vector_type(8)));
typedef float    v2f  __attribute__((ext_vector_type(2)));

#define S_DIM 128
#define CH 64
#define KP 72                 // padded K stride in halfs (144B)
#define NWAVE 4
#define NBLK 36               // 8x8 block grid, lower triangle: 8*9/2
#define TPAD 20               // tbuf row stride in floats (80B; rows 16B-aligned, cols conflict-free)

struct alignas(16) SharedBlk {
    float    tbuf[CH * 16 * TPAD];     // final 64ch x 16 x 16 tile, [ch][li][lj] padded
    _Float16 W2f[CH * KP];             // layer2 weights f16 [M][K] padded
    _Float16 W3f[CH * KP];             // layer3 weights f16
    _Float16 hbuf[NWAVE][16 * KP];     // per-wave activation tile [N][K] padded
    float    fbuf[NWAVE][16 * 4];      // per-wave raw features [col][k]
    float    W1f[CH * 4];              // BN0-folded layer1 weights [o][c] f32
    float    sc1[CH], sh1[CH];
    float    sc2[CH], sh2[CH];
    float    sc3[CH], sh3[CH];
    float    pI[7][16], pJ[7][16];     // px,py,pz,e,pt,rap,phi per particle
};

// tanh-form GELU: x * sigmoid-like, error ~1e-3 (matches f16 WMMA precision), branchless.
// gelu(x) = x / (1 + exp(-2*0.7978845608*(x + 0.044715 x^3)))
__device__ __forceinline__ float gelu_fast(float x) {
    float x2  = x * x;
    float u   = x * fmaf(x2, 0.044715f, 1.0f);
    float e   = __builtin_amdgcn_exp2f(-2.3022077f * u);   // exp2(log2e * -1.59577*u)
    return x * __builtin_amdgcn_rcpf(1.0f + e);
}

__device__ __forceinline__ void decode_tri(int t, int& i, int& j) {
    // t = i*(i+1)/2 + j, j <= i
    float ft = (float)(8 * t + 1);
    int ii = (int)((sqrtf(ft) - 1.0f) * 0.5f);
    while ((ii + 1) * (ii + 2) / 2 <= t) ++ii;
    while (ii * (ii + 1) / 2 > t) --ii;
    i = ii;
    j = t - ii * (ii + 1) / 2;
}

__device__ __forceinline__ v16h load_frag16(const _Float16* p0, const _Float16* p1) {
    v8h lo = *reinterpret_cast<const v8h*>(p0);
    v8h hi = *reinterpret_cast<const v8h*>(p1);
    return __builtin_shufflevector(lo, hi, 0,1,2,3,4,5,6,7,8,9,10,11,12,13,14,15);
}

__global__ __launch_bounds__(128)
void pairembed_fused_kernel(
    const float* __restrict__ x,
    const float* __restrict__ g0,  const float* __restrict__ be0,
    const float* __restrict__ mu0, const float* __restrict__ va0,
    const float* __restrict__ W1,  const float* __restrict__ cb1,
    const float* __restrict__ g1,  const float* __restrict__ be1,
    const float* __restrict__ mu1, const float* __restrict__ va1,
    const float* __restrict__ W2,  const float* __restrict__ cb2,
    const float* __restrict__ g2,  const float* __restrict__ be2,
    const float* __restrict__ mu2, const float* __restrict__ va2,
    const float* __restrict__ W3,  const float* __restrict__ cb3,
    const float* __restrict__ g3,  const float* __restrict__ be3,
    const float* __restrict__ mu3, const float* __restrict__ va3,
    float* __restrict__ y)
{
    __shared__ SharedBlk sm;

    const int tid  = threadIdx.x;
    const int lane = tid & 31;
    const int wave = tid >> 5;
    const int bidx = blockIdx.y;

    int ib, jb;
    decode_tri((int)blockIdx.x, ib, jb);   // lower-triangle 16x16 block coords
    const int i0 = ib * 16;
    const int j0 = jb * 16;

    // ---------------- stage weights / fused params ----------------
    for (int idx = tid; idx < CH * CH; idx += 128) {
        int r = idx >> 6, c = idx & 63;
        sm.W2f[r * KP + c] = (_Float16)W2[idx];
        sm.W3f[r * KP + c] = (_Float16)W3[idx];
    }
    if (tid < CH) {
        // BN0 fold (tiny, recomputed per thread)
        float s0[4], t0[4];
        #pragma unroll
        for (int c = 0; c < 4; ++c) {
            s0[c] = g0[c] * rsqrtf(va0[c] + 1e-5f);
            t0[c] = be0[c] - mu0[c] * s0[c];
        }
        float dot0 = 0.0f;
        #pragma unroll
        for (int c = 0; c < 4; ++c) {
            float w = W1[tid * 4 + c];
            sm.W1f[tid * 4 + c] = w * s0[c];
            dot0 += w * t0[c];
        }
        float s1 = g1[tid] * rsqrtf(va1[tid] + 1e-5f);
        sm.sc1[tid] = s1;
        sm.sh1[tid] = (dot0 + cb1[tid] - mu1[tid]) * s1 + be1[tid];
        float s2 = g2[tid] * rsqrtf(va2[tid] + 1e-5f);
        sm.sc2[tid] = s2;
        sm.sh2[tid] = (cb2[tid] - mu2[tid]) * s2 + be2[tid];
        float s3 = g3[tid] * rsqrtf(va3[tid] + 1e-5f);
        sm.sc3[tid] = s3;
        sm.sh3[tid] = (cb3[tid] - mu3[tid]) * s3 + be3[tid];
    }
    // per-particle kinematics, once per block (16 I-side + 16 J-side)
    if (tid < 32) {
        int side = tid >> 4, t = tid & 15;
        int part = side ? (j0 + t) : (i0 + t);
        const float* xb = x + (size_t)bidx * 4 * S_DIM;
        float px = xb[part];
        float py = xb[S_DIM + part];
        float pz = xb[2 * S_DIM + part];
        float e  = xb[3 * S_DIM + part];
        float pt  = sqrtf(px * px + py * py);
        float rap = 0.5f * log1pf(2.0f * pz / fmaxf(e - pz, 1e-20f));
        float phi = atan2f(py, px);
        float* dst = side ? &sm.pJ[0][0] : &sm.pI[0][0];
        dst[0 * 16 + t] = px;  dst[1 * 16 + t] = py;
        dst[2 * 16 + t] = pz;  dst[3 * 16 + t] = e;
        dst[4 * 16 + t] = pt;  dst[5 * 16 + t] = rap;
        dst[6 * 16 + t] = phi;
    }
    __syncthreads();

    const int col   = lane & 15;   // WMMA column index
    const int khalf = lane >> 4;   // K-half select per ISA fragment layout
    const v8f zf = {0.f,0.f,0.f,0.f,0.f,0.f,0.f,0.f};

    _Float16* hb = sm.hbuf[wave];
    float*    fb = sm.fbuf[wave];

    // each wave processes 4 local rows li; each N-tile = 16 columns lj of one row
    for (int s = 0; s < 4; ++s) {
        const int li = wave * 4 + s;

        // ---- raw pairwise features for (i0+li, j0+col), lanes 0..15 ----
        if (lane < 16) {
            float pxi = sm.pI[0][li],   pyi = sm.pI[1][li];
            float pzi = sm.pI[2][li],   ei  = sm.pI[3][li];
            float pti = sm.pI[4][li],   rapi = sm.pI[5][li], phii = sm.pI[6][li];
            float pxj = sm.pJ[0][lane], pyj = sm.pJ[1][lane];
            float pzj = sm.pJ[2][lane], ej  = sm.pJ[3][lane];
            float ptj = sm.pJ[4][lane], rapj = sm.pJ[5][lane], phij = sm.pJ[6][lane];

            const float PI_F = 3.14159265358979323846f;
            float dphi = fmodf(phii - phij + PI_F, 2.0f * PI_F);
            if (dphi < 0.0f) dphi += 2.0f * PI_F;      // python-style mod
            dphi -= PI_F;
            float drap    = rapi - rapj;
            float delta   = sqrtf(drap * drap + dphi * dphi);
            float lndelta = logf(fmaxf(delta, 1e-8f));
            float ptmin   = fminf(pti, ptj);
            float lnkt    = logf(fmaxf(ptmin * delta, 1e-8f));
            float lnz     = logf(fmaxf(ptmin / fmaxf(pti + ptj, 1e-8f), 1e-8f));
            float sx = pxi + pxj, sy = pyi + pyj, sz = pzi + pzj, se = ei + ej;
            float lnm2 = logf(fmaxf(se*se - sx*sx - sy*sy - sz*sz, 1e-8f));

            *reinterpret_cast<float4*>(fb + lane * 4) =
                make_float4(lnkt, lnz, lndelta, lnm2);
        }
        __syncthreads();

        // ---- layer 1: 4->64 on the matrix pipe (f32 WMMA, K=4) ----
        v8f a1[4];
        {
            v2f bfrag = *reinterpret_cast<const v2f*>(fb + col * 4 + khalf * 2);
            #pragma unroll
            for (int m = 0; m < 4; ++m) {
                v2f afrag = *reinterpret_cast<const v2f*>(
                    sm.W1f + (m * 16 + col) * 4 + khalf * 2);
                a1[m] = __builtin_amdgcn_wmma_f32_16x16x4_f32(
                    false, afrag, false, bfrag, (short)0, zf, false, false);
            }
        }
        // epilogue: BN1 + GELU -> f16 LDS tile [N][K]
        #pragma unroll
        for (int m = 0; m < 4; ++m) {
            v8h pk;
            #pragma unroll
            for (int r = 0; r < 8; ++r) {
                int ch = m * 16 + khalf * 8 + r;
                float v = a1[m][r] * sm.sc1[ch] + sm.sh1[ch];
                pk[r] = (_Float16)gelu_fast(v);
            }
            *reinterpret_cast<v8h*>(hb + col * KP + m * 16 + khalf * 8) = pk;
        }
        __syncthreads();

        // ---- layer 2: 64x64 GEMM (f16 WMMA) ----
        v8f acc[4];
        #pragma unroll
        for (int m = 0; m < 4; ++m) acc[m] = zf;
        #pragma unroll
        for (int ks = 0; ks < 2; ++ks) {
            const _Float16* bp = hb + col * KP + ks * 32 + khalf * 8;
            v16h bfrag = load_frag16(bp, bp + 16);
            #pragma unroll
            for (int m = 0; m < 4; ++m) {
                const _Float16* ap = sm.W2f + (m * 16 + col) * KP + ks * 32;
                v16h afrag = load_frag16(ap + khalf * 8, ap + 16 + khalf * 8);
                acc[m] = __builtin_amdgcn_wmma_f32_16x16x32_f16(
                    false, afrag, false, bfrag, (short)0, acc[m], false, false);
            }
        }
        __syncthreads();
        // epilogue: BN2 + GELU -> f16 LDS tile
        #pragma unroll
        for (int m = 0; m < 4; ++m) {
            v8h pk;
            #pragma unroll
            for (int r = 0; r < 8; ++r) {
                int ch = m * 16 + khalf * 8 + r;
                float v = acc[m][r] * sm.sc2[ch] + sm.sh2[ch];
                pk[r] = (_Float16)gelu_fast(v);
            }
            *reinterpret_cast<v8h*>(hb + col * KP + m * 16 + khalf * 8) = pk;
        }
        __syncthreads();

        // ---- layer 3: 64x64 GEMM (f16 WMMA) + BN3 -> f32 output tile ----
        v8f acc3[4];
        #pragma unroll
        for (int m = 0; m < 4; ++m) acc3[m] = zf;
        #pragma unroll
        for (int ks = 0; ks < 2; ++ks) {
            const _Float16* bp = hb + col * KP + ks * 32 + khalf * 8;
            v16h bfrag = load_frag16(bp, bp + 16);
            #pragma unroll
            for (int m = 0; m < 4; ++m) {
                const _Float16* ap = sm.W3f + (m * 16 + col) * KP + ks * 32;
                v16h afrag = load_frag16(ap + khalf * 8, ap + 16 + khalf * 8);
                acc3[m] = __builtin_amdgcn_wmma_f32_16x16x32_f16(
                    false, afrag, false, bfrag, (short)0, acc3[m], false, false);
            }
        }
        #pragma unroll
        for (int m = 0; m < 4; ++m) {
            #pragma unroll
            for (int r = 0; r < 8; ++r) {
                int ch = m * 16 + khalf * 8 + r;
                float v = acc3[m][r] * sm.sc3[ch] + sm.sh3[ch];
                sm.tbuf[ch * 16 * TPAD + li * TPAD + col] = v;
            }
        }
        __syncthreads();
    }

    // ---------------- fully-coalesced store of tile + transposed tile ----------
    float* yb = y + (size_t)bidx * CH * S_DIM * S_DIM;

    // main tile: y[b][ch][i0+li][j0 .. j0+15], 64B row segments
    for (int q = tid; q < CH * 16 * 4; q += 128) {
        int ch  = q >> 6;
        int li  = (q >> 2) & 15;
        int lj4 = q & 3;
        const float* src = sm.tbuf + ch * 16 * TPAD + li * TPAD + lj4 * 4;
        float4 v = make_float4(src[0], src[1], src[2], src[3]);
        *reinterpret_cast<float4*>(
            yb + (size_t)ch * S_DIM * S_DIM + (i0 + li) * S_DIM + j0 + lj4 * 4) = v;
    }
    // mirror tile (h is symmetric): y[b][ch][j0+lj][i0 .. i0+15]
    if (ib != jb) {
        for (int q = tid; q < CH * 16 * 4; q += 128) {
            int ch  = q >> 6;
            int lj  = (q >> 2) & 15;
            int li4 = q & 3;
            const float* base = sm.tbuf + ch * 16 * TPAD + lj;
            float4 v = make_float4(base[(li4 * 4 + 0) * TPAD],
                                   base[(li4 * 4 + 1) * TPAD],
                                   base[(li4 * 4 + 2) * TPAD],
                                   base[(li4 * 4 + 3) * TPAD]);
            *reinterpret_cast<float4*>(
                yb + (size_t)ch * S_DIM * S_DIM + (j0 + lj) * S_DIM + i0 + li4 * 4) = v;
        }
    }
}

extern "C" void kernel_launch(void* const* d_in, const int* in_sizes, int n_in,
                              void* d_out, int out_size, void* d_ws, size_t ws_size,
                              hipStream_t stream) {
    (void)in_sizes; (void)n_in; (void)out_size; (void)d_ws; (void)ws_size;
    const float* x   = (const float*)d_in[0];
    const float* g0  = (const float*)d_in[1];
    const float* be0 = (const float*)d_in[2];
    const float* mu0 = (const float*)d_in[3];
    const float* va0 = (const float*)d_in[4];
    const float* W1  = (const float*)d_in[5];
    const float* cb1 = (const float*)d_in[6];
    const float* g1  = (const float*)d_in[7];
    const float* be1 = (const float*)d_in[8];
    const float* mu1 = (const float*)d_in[9];
    const float* va1 = (const float*)d_in[10];
    const float* W2  = (const float*)d_in[11];
    const float* cb2 = (const float*)d_in[12];
    const float* g2  = (const float*)d_in[13];
    const float* be2 = (const float*)d_in[14];
    const float* mu2 = (const float*)d_in[15];
    const float* va2 = (const float*)d_in[16];
    const float* W3  = (const float*)d_in[17];
    const float* cb3 = (const float*)d_in[18];
    const float* g3  = (const float*)d_in[19];
    const float* be3 = (const float*)d_in[20];
    const float* mu3 = (const float*)d_in[21];
    const float* va3 = (const float*)d_in[22];
    float* y = (float*)d_out;

    dim3 grid(NBLK, 64);   // 36 lower-triangle 16x16 blocks x 64 batches
    dim3 block(128);       // 4 waves of 32
    pairembed_fused_kernel<<<grid, block, 0, stream>>>(
        x, g0, be0, mu0, va0,
        W1, cb1, g1, be1, mu1, va1,
        W2, cb2, g2, be2, mu2, va2,
        W3, cb3, g3, be3, mu3, va3,
        y);
}
